// Model_68831145886624
// MI455X (gfx1250) — matrix operand order repaired and checked by reading, not compiled, not run on hardware
//
#include <hip/hip_runtime.h>
#include <math.h>

typedef __bf16 bf16;
typedef __attribute__((ext_vector_type(16))) __bf16 bf16x16;
typedef __attribute__((ext_vector_type(8)))  float  f32x8;

constexpr int B_  = 4;
constexpr int T_  = 2048;
constexpr int D_  = 1024;
constexpr int H_  = 8;
constexpr int HD_ = 128;

// (1/sqrt(HD)) * log2(e): folded into the Q projection so attention uses raw
// v_exp_f32 (exp2) with no per-element scaling.
#define QSCALE 0.12751741f

__device__ inline f32x8 zero8() {
  f32x8 z = {0.f, 0.f, 0.f, 0.f, 0.f, 0.f, 0.f, 0.f};
  return z;
}

__device__ inline f32x8 wmma_bf16(bf16x16 a, bf16x16 b, f32x8 c) {
  // 8 args: (neg_a, A, neg_b, B, c_mod, C, reuse_a, reuse_b)
  return __builtin_amdgcn_wmma_f32_16x16x32_bf16(false, a, false, b, (short)0, c,
                                                 false, false);
}

// ---- DPP16 16-lane butterfly all-reduce (pure VALU, no DS traffic) --------
// quad_perm xor1 = 0xB1, quad_perm xor2 = 0x4E, row_ror:4 = 0x124,
// row_ror:8 = 0x128.  After quad stage every lane holds its quad partial;
// the two rotations accumulate all four quads of the 16-lane row.
template <int CTRL>
__device__ inline float dpp_movf(float x) {
  return __builtin_bit_cast(
      float, __builtin_amdgcn_mov_dpp(__builtin_bit_cast(int, x), CTRL, 0xF,
                                      0xF, true));
}
__device__ inline float red_max16(float x) {
  x = fmaxf(x, dpp_movf<0xB1>(x));
  x = fmaxf(x, dpp_movf<0x4E>(x));
  x = fmaxf(x, dpp_movf<0x124>(x));
  x = fmaxf(x, dpp_movf<0x128>(x));
  return x;
}
__device__ inline float red_sum16(float x) {
  x += dpp_movf<0xB1>(x);
  x += dpp_movf<0x4E>(x);
  x += dpp_movf<0x124>(x);
  x += dpp_movf<0x128>(x);
  return x;
}

// A-fragment (16x32 bf16): lane holds row M=lane%16; element i -> K =
// (i%8) + (i/8)*16 + (lane/16)*8.  Caller passes p = row + k0 + (lane/16)*8;
// two contiguous 16B chunks at p and p+16.
__device__ inline bf16x16 ldfragA(const bf16* p) {
  bf16x16 r;
  uint4* u = reinterpret_cast<uint4*>(&r);
  u[0] = *reinterpret_cast<const uint4*>(p);
  u[1] = *reinterpret_cast<const uint4*>(p + 16);
  return r;
}

// B-fragment (32x16 bf16): lane holds column N=lane%16; element i -> K =
// (i%8) + (i/8)*16 + (lane/16)*8.  Caller passes p = col_base + k0 + (lane/16)*8;
// two contiguous 16B chunks at p and p+16.
__device__ inline bf16x16 ldfragB(const bf16* p) {
  bf16x16 r;
  uint4* u = reinterpret_cast<uint4*>(&r);
  u[0] = *reinterpret_cast<const uint4*>(p);
  u[1] = *reinterpret_cast<const uint4*>(p + 16);
  return r;
}

// ---------------------------------------------------------------------------
// fp32 -> bf16 elementwise conversion (vectorized x4)
// ---------------------------------------------------------------------------
__global__ void cvt_f32_to_bf16(const float* __restrict__ in,
                                bf16* __restrict__ out, int n4) {
  int i = blockIdx.x * blockDim.x + threadIdx.x;
  if (i >= n4) return;
  float4 v = reinterpret_cast<const float4*>(in)[i];
  union { bf16 o[4]; uint2 u; } pk;
  pk.o[0] = (bf16)v.x; pk.o[1] = (bf16)v.y;
  pk.o[2] = (bf16)v.z; pk.o[3] = (bf16)v.w;
  reinterpret_cast<uint2*>(out)[i] = pk.u;
}

// ---------------------------------------------------------------------------
// GEMM: out[m,n] = (sum_k A[m,k] * W[n,k] + bias[n]) * oscale
//   A: [M=B*T, K=D] bf16 row-major, W: [N=D, K=D] bf16 row-major.
//   MODE 0: out bf16 head-split [B,H,T,HD]            (Q, K projections)
//   MODE 2: out bf16 head-split transposed [B,H,HD,T] (V projection)
//   MODE 1: out fp32 row-major [M,N]                  (final O projection)
// Block: 256 threads (8 waves as 4(M) x 2(N)); block tile 256x128; wave 64x64.
// ---------------------------------------------------------------------------
template <int MODE>
__global__ __launch_bounds__(256) void gemm_wmma(const bf16* __restrict__ A,
                                                 const bf16* __restrict__ W,
                                                 const float* __restrict__ bias,
                                                 void* __restrict__ outp,
                                                 float oscale) {
  constexpr int K = D_;
  constexpr int N = D_;
  const int lane = threadIdx.x & 31;
  const int wid  = threadIdx.x >> 5;
  const int wm   = wid >> 1;           // 0..3
  const int wn   = wid & 1;            // 0..1
  const int l16  = lane & 15;
  const int lh   = lane >> 4;
  const int mBase = blockIdx.y * 256 + wm * 64;
  const int nBase = blockIdx.x * 128 + wn * 64;

  f32x8 acc[4][4];
#pragma unroll
  for (int i = 0; i < 4; ++i)
#pragma unroll
    for (int j = 0; j < 4; ++j) acc[i][j] = zero8();

  const bf16* aP[4];
  const bf16* bP[4];
#pragma unroll
  for (int i = 0; i < 4; ++i)
    aP[i] = A + (size_t)(mBase + 16 * i + l16) * K + lh * 8;
#pragma unroll
  for (int j = 0; j < 4; ++j)
    bP[j] = W + (size_t)(nBase + 16 * j + l16) * K + lh * 8;

#pragma unroll 2
  for (int k0 = 0; k0 < K; k0 += 32) {
    bf16x16 af[4], bfm[4];
#pragma unroll
    for (int i = 0; i < 4; ++i) af[i] = ldfragA(aP[i] + k0);
#pragma unroll
    for (int j = 0; j < 4; ++j) bfm[j] = ldfragB(bP[j] + k0);
#pragma unroll
    for (int i = 0; i < 4; ++i)
#pragma unroll
      for (int j = 0; j < 4; ++j)
        acc[i][j] = wmma_bf16(af[i], bfm[j], acc[i][j]);
  }

  // Epilogue: C/D layout -> c[r]: M = r + 8*(lane/16), N = lane%16
#pragma unroll
  for (int i = 0; i < 4; ++i)
#pragma unroll
    for (int j = 0; j < 4; ++j)
#pragma unroll
      for (int r = 0; r < 8; ++r) {
        int m = mBase + 16 * i + r + 8 * lh;
        int n = nBase + 16 * j + l16;
        float v = (acc[i][j][r] + bias[n]) * oscale;
        if constexpr (MODE == 1) {
          reinterpret_cast<float*>(outp)[(size_t)m * N + n] = v;
        } else {
          int b = m / T_, t = m % T_;
          int h = n / HD_, d = n % HD_;
          if constexpr (MODE == 0)
            reinterpret_cast<bf16*>(outp)[(((size_t)b * H_ + h) * T_ + t) * HD_ + d] =
                (bf16)v;
          else
            reinterpret_cast<bf16*>(outp)[(((size_t)b * H_ + h) * HD_ + d) * T_ + t] =
                (bf16)v;
        }
      }
}

// ---------------------------------------------------------------------------
// Causal flash attention (exp2 domain; scale pre-folded into Q).
//   Q, Kmat: [B,H,T,HD] bf16.  Vt: [B,H,HD,T] bf16 (pre-transposed).
//   Ctx out: [B,T,D] bf16 (heads re-merged).
// Block 256 threads = 8 waves; each wave owns 16 q rows (block covers 128).
// kv loop split: (nB-1) unmasked full 32-blocks + 1 masked diagonal block.
// grid = (T/128, B*H).
// ---------------------------------------------------------------------------
__global__ __launch_bounds__(256) void attn_kernel(const bf16* __restrict__ Q,
                                                   const bf16* __restrict__ Kmat,
                                                   const bf16* __restrict__ Vt,
                                                   bf16* __restrict__ Ctx) {
  __shared__ bf16 lds[8][16][32];  // per-wave P staging (C-layout -> A-layout)
  const int lane = threadIdx.x & 31;
  const int wave = threadIdx.x >> 5;
  const int l16  = lane & 15;
  const int lh   = lane >> 4;
  const int bh = blockIdx.y;        // 0..B*H-1
  const int b  = bh >> 3;
  const int h  = bh & 7;
  const int qBase = blockIdx.x * 128 + wave * 16;

  const bf16* Qb = Q    + (size_t)bh * T_ * HD_;
  const bf16* Kb = Kmat + (size_t)bh * T_ * HD_;
  const bf16* Vb = Vt   + (size_t)bh * HD_ * T_;

  // Preload Q A-fragments for the full HD=128 (4 chunks of K=32)
  bf16x16 qa[4];
#pragma unroll
  for (int ks = 0; ks < 4; ++ks)
    qa[ks] = ldfragA(Qb + (size_t)(qBase + l16) * HD_ + ks * 32 + lh * 8);

  f32x8 o[8];
  float mr[8], lr[8];
#pragma unroll
  for (int j = 0; j < 8; ++j) o[j] = zero8();
#pragma unroll
  for (int r = 0; r < 8; ++r) { mr[r] = -INFINITY; lr[r] = 0.0f; }

  // One kv step of 32 columns; MASKED only for the diagonal block.
  auto kv_step = [&](int kv0, bool masked) {
    // ---- S = Q @ K^T (logits already in exp2 domain via Q pre-scale) ----
    f32x8 s0 = zero8(), s1 = zero8();
#pragma unroll
    for (int ks = 0; ks < 4; ++ks) {
      bf16x16 k0f = ldfragB(Kb + (size_t)(kv0 + l16) * HD_ + ks * 32 + lh * 8);
      bf16x16 k1f = ldfragB(Kb + (size_t)(kv0 + 16 + l16) * HD_ + ks * 32 + lh * 8);
      s0 = wmma_bf16(qa[ks], k0f, s0);
      s1 = wmma_bf16(qa[ks], k1f, s1);
    }

    // ---- online softmax (row groups = 16-lane halves, DPP reductions) ----
#pragma unroll
    for (int r = 0; r < 8; ++r) {
      float v0 = s0[r];
      float v1 = s1[r];
      if (masked) {
        const int qr = qBase + r + 8 * lh;
        const int c0 = kv0 + l16;
        v0 = (c0      <= qr) ? v0 : -INFINITY;
        v1 = (c0 + 16 <= qr) ? v1 : -INFINITY;
      }
      float rm  = red_max16(fmaxf(v0, v1));
      float nm  = fmaxf(mr[r], rm);
      float fac = __builtin_amdgcn_exp2f(mr[r] - nm);
      float p0  = __builtin_amdgcn_exp2f(v0 - nm);
      float p1  = __builtin_amdgcn_exp2f(v1 - nm);
      float rs  = red_sum16(p0 + p1);
      lr[r] = lr[r] * fac + rs;
      mr[r] = nm;
#pragma unroll
      for (int j = 0; j < 8; ++j) o[j][r] = o[j][r] * fac;
      // stage P tile (16 rows x 32 kv) into this wave's LDS slab
      lds[wave][r + 8 * lh][l16]      = (bf16)p0;
      lds[wave][r + 8 * lh][l16 + 16] = (bf16)p1;
    }
    asm volatile("s_wait_dscnt 0x0" ::: "memory");

    // reload P as an A-fragment (16x32, K = kv); same-wave DS ops stay ordered
    bf16x16 pa;
    {
      const bf16* lp = &lds[wave][l16][0];
      uint4* u = reinterpret_cast<uint4*>(&pa);
      u[0] = *reinterpret_cast<const uint4*>(lp + lh * 8);
      u[1] = *reinterpret_cast<const uint4*>(lp + 16 + lh * 8);
    }

    // ---- O += P @ V : 8 column tiles of HD ----
#pragma unroll
    for (int j = 0; j < 8; ++j) {
      bf16x16 vf = ldfragB(Vb + (size_t)(j * 16 + l16) * T_ + kv0 + lh * 8);
      o[j] = wmma_bf16(pa, vf, o[j]);
    }
  };

  // Causal bound: wave needs kv < qBase+16.  Exactly one 32-wide block
  // straddles the diagonal for 16-aligned qBase; all earlier blocks are full.
  const int nB = (qBase + 47) >> 5;  // total 32-wide kv blocks
  for (int ib = 0; ib < nB - 1; ++ib) kv_step(ib * 32, false);
  kv_step((nB - 1) * 32, true);

  // ---- normalize and write ctx [B,T,D] with heads merged ----
#pragma unroll
  for (int r = 0; r < 8; ++r) {
    float inv = 1.0f / lr[r];
    int t = qBase + r + 8 * lh;
#pragma unroll
    for (int j = 0; j < 8; ++j) {
      int col = h * HD_ + j * 16 + l16;
      Ctx[((size_t)b * T_ + t) * D_ + col] = (bf16)(o[j][r] * inv);
    }
  }
}

// ---------------------------------------------------------------------------
// Launch pipeline
// ---------------------------------------------------------------------------
extern "C" void kernel_launch(void* const* d_in, const int* in_sizes, int n_in,
                              void* d_out, int out_size, void* d_ws,
                              size_t ws_size, hipStream_t stream) {
  (void)in_sizes; (void)n_in; (void)out_size; (void)ws_size;

  const float* x  = (const float*)d_in[0];
  const float* y  = (const float*)d_in[1];
  const float* Wq = (const float*)d_in[2];
  const float* bq = (const float*)d_in[3];
  const float* Wk = (const float*)d_in[4];
  const float* bk = (const float*)d_in[5];
  const float* Wv = (const float*)d_in[6];
  const float* bv = (const float*)d_in[7];
  const float* Wo = (const float*)d_in[8];
  const float* bo = (const float*)d_in[9];

  const size_t BTD = (size_t)B_ * T_ * D_;  // 8,388,608 elems
  const size_t DD  = (size_t)D_ * D_;       // 1,048,576 elems

  char* ws = (char*)d_ws;
  size_t off = 0;
  bf16* Xh  = (bf16*)(ws + off); off += BTD * 2;
  bf16* Yh  = (bf16*)(ws + off); off += BTD * 2;
  bf16* Whq = (bf16*)(ws + off); off += DD * 2;
  bf16* Whk = (bf16*)(ws + off); off += DD * 2;
  bf16* Whv = (bf16*)(ws + off); off += DD * 2;
  bf16* Who = (bf16*)(ws + off); off += DD * 2;
  bf16* Qh  = (bf16*)(ws + off); off += BTD * 2;
  bf16* Kh  = (bf16*)(ws + off); off += BTD * 2;
  bf16* Vth = (bf16*)(ws + off); off += BTD * 2;
  bf16* Ctx = (bf16*)(ws + off); off += BTD * 2;  // ~104 MB total

  // 1) fp32 -> bf16 conversions
  {
    int n4 = (int)(BTD / 4);
    cvt_f32_to_bf16<<<(n4 + 255) / 256, 256, 0, stream>>>(x, Xh, n4);
    cvt_f32_to_bf16<<<(n4 + 255) / 256, 256, 0, stream>>>(y, Yh, n4);
    int w4 = (int)(DD / 4);
    cvt_f32_to_bf16<<<(w4 + 255) / 256, 256, 0, stream>>>(Wq, Whq, w4);
    cvt_f32_to_bf16<<<(w4 + 255) / 256, 256, 0, stream>>>(Wk, Whk, w4);
    cvt_f32_to_bf16<<<(w4 + 255) / 256, 256, 0, stream>>>(Wv, Whv, w4);
    cvt_f32_to_bf16<<<(w4 + 255) / 256, 256, 0, stream>>>(Wo, Who, w4);
  }

  // 2) projections: grid = (N/128, M/256) = (8, 32)
  dim3 gGemm(D_ / 128, (B_ * T_) / 256);
  gemm_wmma<0><<<gGemm, 256, 0, stream>>>(Xh, Whq, bq, Qh, QSCALE);  // Q (pre-scaled)
  gemm_wmma<0><<<gGemm, 256, 0, stream>>>(Yh, Whk, bk, Kh, 1.0f);   // K
  gemm_wmma<2><<<gGemm, 256, 0, stream>>>(Yh, Whv, bv, Vth, 1.0f);  // V^T

  // 3) causal flash attention: grid = (T/128, B*H) = (16, 32)
  attn_kernel<<<dim3(T_ / 128, B_ * H_), 256, 0, stream>>>(Qh, Kh, Vth, Ctx);

  // 4) output projection -> fp32 d_out [B,T,D]
  gemm_wmma<1><<<gGemm, 256, 0, stream>>>(Ctx, Who, bo, (float*)d_out, 1.0f);
}
